// TransformerModel_6322191860452
// MI455X (gfx1250) — matrix operand order repaired and checked by reading, not compiled, not run on hardware
//
#include <hip/hip_runtime.h>

// ---------------------------------------------------------------------------
// Transformer encoder layer for MI455X (gfx1250, wave32, WMMA bf16)
//   B=1, S=4096, D=1024, H=16, DH=64
// Async-DMA (global_load_async_to_lds) double-buffered staging throughout.
// ---------------------------------------------------------------------------

#define S_LEN 4096
#define D_MOD 1024
#define N_HEAD 16
#define D_HEAD 64

typedef __attribute__((ext_vector_type(16))) __bf16 v16bf;
typedef __attribute__((ext_vector_type(8)))  __bf16 v8bf;
typedef __attribute__((ext_vector_type(8)))  float  v8f;

// A 16x32 bf16 fragment's 16 lane-elements are two contiguous 8-element runs:
//   e=0..7  -> K = half*8 + e ;  e=8..15 -> K = 16 + half*8 + (e-8)
// (CDNA5 ISA 7.12.2, "16-bit A-Matrix 16x32")
union F16 { v16bf v; v8bf h[2]; };

__device__ __forceinline__ v8f wmma_bf16(v16bf a, v16bf b, v8f c) {
    return __builtin_amdgcn_wmma_f32_16x16x32_bf16(
        false, a, false, b, (short)0, c, false, false);
}

// 16B async DMA: global -> LDS, tracked by ASYNCcnt (CDNA5 §15.18.3 op 98)
__device__ __forceinline__ void async_ld16(unsigned ldsDst, const __bf16* gsrc) {
    asm volatile("global_load_async_to_lds_b128 %0, %1, off"
                 :: "v"(ldsDst), "v"((unsigned long long)(uintptr_t)gsrc)
                 : "memory");
}
__device__ __forceinline__ void wait_async0() {
    asm volatile("s_wait_asynccnt 0x0" ::: "memory");
}

// ---------------------------------------------------------------------------
// 1) h = x + positional encoding ; emit f32 (residual) and bf16 (GEMM A)
// ---------------------------------------------------------------------------
__global__ __launch_bounds__(256) void pe_kernel(const float* __restrict__ x,
                                                 float* __restrict__ h32,
                                                 __bf16* __restrict__ hbf) {
    int idx = blockIdx.x * 256 + threadIdx.x;      // over S*D
    int s = idx >> 10;
    int d = idx & 1023;
    float div = __expf(-9.210340371976184f * (float)(2 * (d >> 1)) / 1024.0f);
    float ang = (float)s * div;
    float pe  = (d & 1) ? __cosf(ang) : __sinf(ang);
    float v = x[idx] + pe;
    h32[idx] = v;
    hbf[idx] = (__bf16)v;
}

// ---------------------------------------------------------------------------
// 2) weight conversion with transpose: W[K,N] f32 -> Wt[N,K] bf16.
// ---------------------------------------------------------------------------
__global__ __launch_bounds__(256) void cvt_tr_kernel(const float* __restrict__ W,
                                                     __bf16* __restrict__ Wt) {
    int idx = blockIdx.x * 256 + threadIdx.x;  // over N*K (1024*1024)
    int n = idx >> 10;
    int k = idx & 1023;
    Wt[idx] = (__bf16)W[(size_t)k * D_MOD + n];
}

// ---------------------------------------------------------------------------
// 3) GEMM: C[M,N] = A[M,K](bf16) @ W[K,N](bf16) + bias, W given as Bt[N,K].
//    Block tile 128x128x32; 8 waves (2x4); wave = 64x32 = 4x2 WMMA tiles.
//    Double-buffered LDS fed by async DMA; each thread owns 2 fixed 16B
//    chunks per operand; WMMAs overlap the next tile's DMA.
// ---------------------------------------------------------------------------
__global__ __launch_bounds__(256)
void gemm_bf16_kernel(const __bf16* __restrict__ A,   // [M,K]
                      const __bf16* __restrict__ Bt,  // [N,K]
                      const float*  __restrict__ bias,// [N]
                      const float*  __restrict__ resid, // [M,N] or null
                      float*  __restrict__ Cf,          // [M,N] or null
                      __bf16* __restrict__ Cbf,         // [M,N] or null
                      __bf16* __restrict__ Cbt,         // [N,M] or null
                      int M, int N, int K) {
    __shared__ __attribute__((aligned(32))) __bf16 As[2][8][32][16]; // 2 x 8KB
    __shared__ __attribute__((aligned(32))) __bf16 Bs[2][8][32][16]; // 2 x 8KB

    const int tid  = threadIdx.x;
    const int lane = tid & 31;
    const int wave = tid >> 5;
    const int mBlock = blockIdx.y * 128;
    const int nBlock = blockIdx.x * 128;
    const int waveM = (wave >> 2) * 64;
    const int waveN = (wave & 3) * 32;

    // This thread's two staging chunks (of 512): chunk i -> row i>>2, 8-elem c=i&3
    const int r0 = tid >> 2,         c0 = tid & 3;
    const int r1 = (tid + 256) >> 2, c1 = (tid + 256) & 3;
    const __bf16* aS0 = A  + (size_t)(mBlock + r0) * K + c0 * 8;
    const __bf16* aS1 = A  + (size_t)(mBlock + r1) * K + c1 * 8;
    const __bf16* bS0 = Bt + (size_t)(nBlock + r0) * K + c0 * 8;
    const __bf16* bS1 = Bt + (size_t)(nBlock + r1) * K + c1 * 8;
    // A-chunk dest: half = c&1, elem off = (c>>1)*8 ; B-chunk: same
    unsigned aD0[2], aD1[2], bD0[2], bD1[2];
#pragma unroll
    for (int b = 0; b < 2; ++b) {
        aD0[b] = (unsigned)(uintptr_t)&As[b][r0 >> 4][(r0 & 15) + 16 * (c0 & 1)][(c0 >> 1) * 8];
        aD1[b] = (unsigned)(uintptr_t)&As[b][r1 >> 4][(r1 & 15) + 16 * (c1 & 1)][(c1 >> 1) * 8];
        bD0[b] = (unsigned)(uintptr_t)&Bs[b][r0 >> 4][(r0 & 15) + 16 * (c0 & 1)][(c0 >> 1) * 8];
        bD1[b] = (unsigned)(uintptr_t)&Bs[b][r1 >> 4][(r1 & 15) + 16 * (c1 & 1)][(c1 >> 1) * 8];
    }

    v8f acc[4][2];
#pragma unroll
    for (int i = 0; i < 4; ++i)
#pragma unroll
        for (int j = 0; j < 2; ++j)
#pragma unroll
            for (int r = 0; r < 8; ++r) acc[i][j][r] = 0.0f;

    auto stage = [&](int b) {
        async_ld16(aD0[b], aS0); async_ld16(aD1[b], aS1);
        async_ld16(bD0[b], bS0); async_ld16(bD1[b], bS1);
        aS0 += 32; aS1 += 32; bS0 += 32; bS1 += 32;   // next 32-K tile
    };
    auto compute = [&](int b) {
        v16bf af[4], bfr[2];
#pragma unroll
        for (int i = 0; i < 4; ++i)
            af[i] = *(const v16bf*)&As[b][(waveM >> 4) + i][lane][0];
#pragma unroll
        for (int j = 0; j < 2; ++j)
            bfr[j] = *(const v16bf*)&Bs[b][(waveN >> 4) + j][lane][0];
#pragma unroll
        for (int i = 0; i < 4; ++i)
#pragma unroll
            for (int j = 0; j < 2; ++j)
                acc[i][j] = wmma_bf16(af[i], bfr[j], acc[i][j]);
    };

    const int nT = K / 32;   // 32 tiles (even)
    stage(0);
    wait_async0();
    __syncthreads();
    for (int t = 0; t < nT; t += 2) {
        stage(1);                       // t+1 < nT always (nT even)
        compute(0);
        wait_async0();
        __syncthreads();
        if (t + 2 < nT) stage(0);
        compute(1);
        wait_async0();
        __syncthreads();
    }

    // ---- epilogue ----
    const int half = lane >> 4;
#pragma unroll
    for (int i = 0; i < 4; ++i) {
#pragma unroll
        for (int j = 0; j < 2; ++j) {
            int n = nBlock + waveN + j * 16 + (lane & 15);
            float bv = bias ? bias[n] : 0.0f;
            int m0 = mBlock + waveM + i * 16 + 8 * half;  // 8 consecutive rows
            if (Cbt) {  // transposed store: one contiguous 16B chunk per lane
                union { v8bf v; __bf16 e[8]; } u;
#pragma unroll
                for (int r = 0; r < 8; ++r) u.e[r] = (__bf16)(acc[i][j][r] + bv);
                *(v8bf*)&Cbt[(size_t)n * M + m0] = u.v;
            }
            if (Cf || Cbf) {
#pragma unroll
                for (int r = 0; r < 8; ++r) {
                    int m = m0 + r;
                    float v = acc[i][j][r] + bv;
                    if (resid) v += resid[(size_t)m * N + n];
                    if (Cf)  Cf[(size_t)m * N + n]  = v;
                    if (Cbf) Cbf[(size_t)m * N + n] = (__bf16)v;
                }
            }
        }
    }
}

// ---------------------------------------------------------------------------
// 4) Flash attention. Grid (S/128, H), 256 threads = 8 waves; each wave owns
//    16 query rows. KV tiles (64 wide) double-buffered via async DMA; V comes
//    from the pre-transposed Vt[D,S] so all fragments are contiguous v16bf.
// ---------------------------------------------------------------------------
__global__ __launch_bounds__(256)
void attn_kernel(const __bf16* __restrict__ Q,   // [S, D] head-interleaved
                 const __bf16* __restrict__ K,   // [S, D]
                 const __bf16* __restrict__ Vt,  // [D, S] transposed
                 __bf16* __restrict__ O) {
    __shared__ __attribute__((aligned(32))) __bf16 kt[2][64][64];     // [kv][dim]
    __shared__ __attribute__((aligned(32))) __bf16 vtT[2][64][64];    // [dim][kv]
    __shared__ __attribute__((aligned(32))) __bf16 pstage[8][16][64]; // per-wave P

    const int h     = blockIdx.y;
    const int qBase = blockIdx.x * 128;
    const int tid   = threadIdx.x;
    const int lane  = tid & 31;
    const int wave  = tid >> 5;
    const int half  = lane >> 4;
    const int hcol  = h * D_HEAD;

    // staging: 512 chunks per operand; this thread owns chunks tid and tid+256
    const int r0 = tid >> 3,         c0 = tid & 7;
    const int r1 = (tid + 256) >> 3, c1 = (tid + 256) & 7;
    const __bf16* kS0 = K  + (size_t)r0 * D_MOD + hcol + c0 * 8;
    const __bf16* kS1 = K  + (size_t)r1 * D_MOD + hcol + c1 * 8;
    const __bf16* vS0 = Vt + (size_t)(hcol + r0) * S_LEN + c0 * 8;
    const __bf16* vS1 = Vt + (size_t)(hcol + r1) * S_LEN + c1 * 8;
    unsigned kD0[2], kD1[2], vD0[2], vD1[2];
#pragma unroll
    for (int b = 0; b < 2; ++b) {
        kD0[b] = (unsigned)(uintptr_t)&kt[b][r0][c0 * 8];
        kD1[b] = (unsigned)(uintptr_t)&kt[b][r1][c1 * 8];
        vD0[b] = (unsigned)(uintptr_t)&vtT[b][r0][c0 * 8];
        vD1[b] = (unsigned)(uintptr_t)&vtT[b][r1][c1 * 8];
    }

    // Q fragments: two contiguous 16B runs per lane per fragment
    v16bf qf[2];
    {
        const __bf16* qp = Q + (size_t)(qBase + wave * 16 + (lane & 15)) * D_MOD + hcol;
#pragma unroll
        for (int ks = 0; ks < 2; ++ks) {
            F16 f;
            f.h[0] = *(const v8bf*)(qp + 32 * ks + half * 16);
            f.h[1] = *(const v8bf*)(qp + 32 * ks + half * 16 + 8);
            qf[ks] = f.v;
        }
    }

    v8f acc[4];
    float mrow[8], lrow[8];
#pragma unroll
    for (int nt = 0; nt < 4; ++nt)
#pragma unroll
        for (int r = 0; r < 8; ++r) acc[nt][r] = 0.0f;
#pragma unroll
    for (int r = 0; r < 8; ++r) { mrow[r] = -3.0e38f; lrow[r] = 0.0f; }

    const float sc = 0.125f;  // 1/sqrt(64)

    auto stage = [&](int b) {
        async_ld16(kD0[b], kS0); async_ld16(kD1[b], kS1);
        async_ld16(vD0[b], vS0); async_ld16(vD1[b], vS1);
        kS0 += (size_t)64 * D_MOD; kS1 += (size_t)64 * D_MOD;  // next 64 kv rows
        vS0 += 64; vS1 += 64;                                   // next 64 kv cols
    };

    auto compute = [&](int b) {
        // scores: S = Q (16x64) @ K^T (64x64)
        v8f s[4];
#pragma unroll
        for (int nt = 0; nt < 4; ++nt) {
#pragma unroll
            for (int r = 0; r < 8; ++r) s[nt][r] = 0.0f;
            int n = (lane & 15) + nt * 16;
#pragma unroll
            for (int ks = 0; ks < 2; ++ks) {
                v16bf bb = *(const v16bf*)&kt[b][n][32 * ks + half * 16];
                s[nt] = wmma_bf16(qf[ks], bb, s[nt]);
            }
        }
        // online softmax
        float mnew[8], scl[8];
#pragma unroll
        for (int r = 0; r < 8; ++r) {
            float t = fmaxf(fmaxf(s[0][r], s[1][r]), fmaxf(s[2][r], s[3][r])) * sc;
#pragma unroll
            for (int off = 1; off < 16; off <<= 1)
                t = fmaxf(t, __shfl_xor(t, off, 32));
            mnew[r] = fmaxf(mrow[r], t);
            scl[r]  = __expf(mrow[r] - mnew[r]);
            mrow[r] = mnew[r];
        }
#pragma unroll
        for (int nt = 0; nt < 4; ++nt)
#pragma unroll
            for (int r = 0; r < 8; ++r)
                s[nt][r] = __expf(s[nt][r] * sc - mnew[r]);
#pragma unroll
        for (int r = 0; r < 8; ++r) {
            float rs = s[0][r] + s[1][r] + s[2][r] + s[3][r];
#pragma unroll
            for (int off = 1; off < 16; off <<= 1)
                rs += __shfl_xor(rs, off, 32);
            lrow[r] = lrow[r] * scl[r] + rs;
#pragma unroll
            for (int nt = 0; nt < 4; ++nt) acc[nt][r] *= scl[r];
        }
        // P (C-order) -> A-fragment order via per-wave LDS
#pragma unroll
        for (int nt = 0; nt < 4; ++nt)
#pragma unroll
            for (int r = 0; r < 8; ++r)
                pstage[wave][r + 8 * half][(lane & 15) + 16 * nt] = (__bf16)s[nt][r];
        // O += P (16x64) @ V (64x64)
#pragma unroll
        for (int ks = 0; ks < 2; ++ks) {
            F16 pf;
            pf.h[0] = *(const v8bf*)&pstage[wave][lane & 15][32 * ks + half * 16];
            pf.h[1] = *(const v8bf*)&pstage[wave][lane & 15][32 * ks + half * 16 + 8];
#pragma unroll
            for (int nt = 0; nt < 4; ++nt) {
                v16bf vb = *(const v16bf*)&vtT[b][(lane & 15) + 16 * nt][32 * ks + half * 16];
                acc[nt] = wmma_bf16(pf.v, vb, acc[nt]);
            }
        }
    };

    const int nT = S_LEN / 64;  // 64 tiles (even)
    stage(0);
    wait_async0();
    __syncthreads();
    for (int t = 0; t < nT; t += 2) {
        stage(1);
        compute(0);
        wait_async0();
        __syncthreads();
        if (t + 2 < nT) stage(0);
        compute(1);
        wait_async0();
        __syncthreads();
    }

    // normalize and write O (bf16, head-interleaved, row-major)
#pragma unroll
    for (int nt = 0; nt < 4; ++nt)
#pragma unroll
        for (int r = 0; r < 8; ++r) {
            int row = qBase + wave * 16 + r + 8 * half;
            int col = (lane & 15) + nt * 16;
            O[(size_t)row * D_MOD + hcol + col] = (__bf16)(acc[nt][r] / lrow[r]);
        }
}

// ---------------------------------------------------------------------------
// 5) LayerNorm over last dim (1024); one block per row.
// ---------------------------------------------------------------------------
__global__ __launch_bounds__(256)
void ln_kernel(const float* __restrict__ y, const float* __restrict__ gamma,
               const float* __restrict__ beta, float* __restrict__ out) {
    __shared__ float ss[8], ss2[8];
    const int row  = blockIdx.x;
    const int lane = threadIdx.x & 31;
    const int wave = threadIdx.x >> 5;
    const float* yr = y + (size_t)row * D_MOD;

    float s = 0.0f, s2 = 0.0f;
    for (int i = threadIdx.x; i < D_MOD; i += 256) {
        float v = yr[i];
        s += v; s2 += v * v;
    }
#pragma unroll
    for (int off = 1; off < 32; off <<= 1) {
        s  += __shfl_xor(s,  off, 32);
        s2 += __shfl_xor(s2, off, 32);
    }
    if (lane == 0) { ss[wave] = s; ss2[wave] = s2; }
    __syncthreads();
    float tot = 0.0f, tot2 = 0.0f;
#pragma unroll
    for (int w = 0; w < 8; ++w) { tot += ss[w]; tot2 += ss2[w]; }
    float mu  = tot * (1.0f / D_MOD);
    float var = tot2 * (1.0f / D_MOD) - mu * mu;
    float inv = rsqrtf(var + 1e-5f);
    for (int i = threadIdx.x; i < D_MOD; i += 256)
        out[(size_t)row * D_MOD + i] = (yr[i] - mu) * inv * gamma[i] + beta[i];
}

// ---------------------------------------------------------------------------
// Launch
// ---------------------------------------------------------------------------
extern "C" void kernel_launch(void* const* d_in, const int* in_sizes, int n_in,
                              void* d_out, int out_size, void* d_ws, size_t ws_size,
                              hipStream_t stream) {
    const float* x     = (const float*)d_in[0];
    const float* Wq    = (const float*)d_in[1];
    const float* bq    = (const float*)d_in[2];
    const float* Wk    = (const float*)d_in[3];
    const float* bk    = (const float*)d_in[4];
    const float* Wv    = (const float*)d_in[5];
    const float* bv    = (const float*)d_in[6];
    const float* Wo    = (const float*)d_in[7];
    const float* bo    = (const float*)d_in[8];
    const float* gamma = (const float*)d_in[9];
    const float* beta  = (const float*)d_in[10];
    float* out = (float*)d_out;

    const size_t SD = (size_t)S_LEN * D_MOD;
    const size_t DD = (size_t)D_MOD * D_MOD;

    char* p = (char*)d_ws;
    auto take = [&](size_t bytes) {
        void* r = (void*)p;
        p += (bytes + 255) & ~(size_t)255;
        return r;
    };
    float*  h32 = (float*)take(SD * 4);
    __bf16* hbf = (__bf16*)take(SD * 2);
    __bf16* wqt = (__bf16*)take(DD * 2);   // transposed bf16 weights [N,K]
    __bf16* wkt = (__bf16*)take(DD * 2);
    __bf16* wvt = (__bf16*)take(DD * 2);
    __bf16* wot = (__bf16*)take(DD * 2);
    __bf16* qb  = (__bf16*)take(SD * 2);   // Q   [S,D]
    __bf16* kb  = (__bf16*)take(SD * 2);   // K   [S,D]
    __bf16* vtg = (__bf16*)take(SD * 2);   // V^T [D,S]
    __bf16* ob  = (__bf16*)take(SD * 2);   // attention out [S,D]
    float*  y32 = (float*)take(SD * 4);

    pe_kernel<<<SD / 256, 256, 0, stream>>>(x, h32, hbf);

    cvt_tr_kernel<<<DD / 256, 256, 0, stream>>>(Wq, wqt);
    cvt_tr_kernel<<<DD / 256, 256, 0, stream>>>(Wk, wkt);
    cvt_tr_kernel<<<DD / 256, 256, 0, stream>>>(Wv, wvt);
    cvt_tr_kernel<<<DD / 256, 256, 0, stream>>>(Wo, wot);

    dim3 ggrid(D_MOD / 128, S_LEN / 128);
    gemm_bf16_kernel<<<ggrid, 256, 0, stream>>>(hbf, wqt, bq, nullptr, nullptr, qb,
                                                nullptr, S_LEN, D_MOD, D_MOD);
    gemm_bf16_kernel<<<ggrid, 256, 0, stream>>>(hbf, wkt, bk, nullptr, nullptr, kb,
                                                nullptr, S_LEN, D_MOD, D_MOD);
    gemm_bf16_kernel<<<ggrid, 256, 0, stream>>>(hbf, wvt, bv, nullptr, nullptr, nullptr,
                                                vtg, S_LEN, D_MOD, D_MOD);

    dim3 agrid(S_LEN / 128, N_HEAD);
    attn_kernel<<<agrid, 256, 0, stream>>>(qb, kb, vtg, ob);

    gemm_bf16_kernel<<<ggrid, 256, 0, stream>>>(ob, wot, bo, h32, y32, nullptr,
                                                nullptr, S_LEN, D_MOD, D_MOD);

    ln_kernel<<<S_LEN, 256, 0, stream>>>(y32, gamma, beta, out);
}
